// SafeDrugModel_mod_64501818851473
// MI455X (gfx1250) — matrix operand order repaired and checked by reading, not compiled, not run on hardware
//
#include <hip/hip_runtime.h>
#include <math.h>

typedef __bf16 bf16_t;
typedef __bf16 v16bf __attribute__((ext_vector_type(16)));
typedef float  v8f   __attribute__((ext_vector_type(8)));

#define DIM   256
#define BATCH 128
#define TLEN  32
#define LDg   48
#define LPg   32
#define H1g   500
#define V2g   200
#define NMOLg 250
#define NAg   8000
#define G3D   768   /* 3*DIM */

static __device__ __forceinline__ float sigmoidf(float x){ return 1.0f/(1.0f+__expf(-x)); }

// ---------------- WMMA tile loaders (contiguous k order: k = 16*(lane/16) + element) ----------------
// 16x32 bf16 A tile: lane L holds row M=L%16; lanes 0-15: K = k0+0..15,
// lanes 16-31: K = k0+16..31.
static __device__ __forceinline__ v16bf load_A16x32_bf(const bf16_t* A, int lda, int row0, int k0){
  int lane = threadIdx.x & 31;
  int m  = row0 + (lane & 15);
  int kb = k0 + ((lane >> 4) << 4);
  const bf16_t* p = A + (size_t)m * lda + kb;
  v16bf a;
#pragma unroll
  for (int e = 0; e < 8; ++e) a[e] = p[e];
#pragma unroll
  for (int e = 0; e < 8; ++e) a[8+e] = p[8+e];
  return a;
}
// Same A tile but source is f32 (adjacency) converted on the fly.
static __device__ __forceinline__ v16bf load_A16x32_f32(const float* A, int lda, int row0, int k0){
  int lane = threadIdx.x & 31;
  int m  = row0 + (lane & 15);
  int kb = k0 + ((lane >> 4) << 4);
  const float* p = A + (size_t)m * lda + kb;
  v16bf a;
#pragma unroll
  for (int e = 0; e < 8; ++e) a[e] = (bf16_t)p[e];
#pragma unroll
  for (int e = 0; e < 8; ++e) a[8+e] = (bf16_t)p[8+e];
  return a;
}
// 32x16 bf16 B tile where memory holds B^T: row n (length ldw) contains all K for col n.
// Works for [N x K] weight matrices (A @ W^T) AND for explicitly transposed activations.
// Lane L holds col N=L%16; lanes 0-15: K=k0+0..15, lanes 16-31: K=k0+16..31.
// Per-lane access is 16 contiguous bf16 (32B) -> lowers to 2x global_load_b128.
static __device__ __forceinline__ v16bf load_B32x16_wt(const bf16_t* W, int ldw, int k0, int col0){
  int lane = threadIdx.x & 31;
  int n  = col0 + (lane & 15);
  int kb = k0 + ((lane >> 4) << 4);
  const bf16_t* p = W + (size_t)n * ldw + kb;
  v16bf b;
#pragma unroll
  for (int e = 0; e < 16; ++e) b[e] = p[e];
  return b;
}
static __device__ __forceinline__ v8f wmma_bf16(v16bf a, v16bf b, v8f c){
  return __builtin_amdgcn_wmma_f32_16x16x32_bf16(false, a, false, b, (short)0, c, false, false);
}

// ---------------- utility kernels ----------------
__global__ void k_zero_f32(float* p, int n){
  for (int i = blockIdx.x*blockDim.x + threadIdx.x; i < n; i += gridDim.x*blockDim.x) p[i] = 0.f;
}
__global__ void k_f32_to_bf16(const float* s, bf16_t* d, int n){
  for (int i = blockIdx.x*blockDim.x + threadIdx.x; i < n; i += gridDim.x*blockDim.x) d[i] = (bf16_t)s[i];
}

// ---------------- embeddings: visit-level pooled sums -> bf16 GRU inputs ----------------
__global__ void k_embed(const int* __restrict__ diag, const int* __restrict__ pro,
                        const float* __restrict__ emb0, const float* __restrict__ emb1,
                        bf16_t* X0, bf16_t* X1){
  int idx = blockIdx.x * blockDim.x + threadIdx.x;   // over BATCH*TLEN*DIM
  int d  = idx & (DIM-1);
  int bt = idx >> 8;
  if (blockIdx.y == 0) {
    const int* ids = diag + bt * LDg;
    float s = 0.f;
    for (int l = 0; l < LDg; ++l) s += emb0[(size_t)ids[l]*DIM + d];
    X0[(size_t)bt*DIM + d] = (bf16_t)s;
  } else {
    const int* ids = pro + bt * LPg;
    float s = 0.f;
    for (int l = 0; l < LPg; ++l) s += emb1[(size_t)ids[l]*DIM + d];
    X1[(size_t)bt*DIM + d] = (bf16_t)s;
  }
}

// ---------------- generic WMMA GEMM: C[MxN] = A[MxK] @ W[NxK]^T + bias ------------------
// Optional relu; optional bf16 row-major copy (Cbf) and/or bf16 TRANSPOSED copy
// (CbfT, laid out [N x M] so it can feed load_B32x16_wt contiguously as a B operand).
__global__ void k_gemm_wt(const bf16_t* __restrict__ A, const bf16_t* __restrict__ W,
                          const float* __restrict__ bias, float* C, bf16_t* Cbf, bf16_t* CbfT,
                          int M, int N, int K, int relu){
  int wave = (blockIdx.x * blockDim.x + threadIdx.x) >> 5;
  int ntiles = N >> 4;
  int mtile = wave / ntiles, ntile = wave % ntiles;
  int row0 = mtile << 4, col0 = ntile << 4;
  v8f acc = {};
  for (int k0 = 0; k0 < K; k0 += 32){
    v16bf a = load_A16x32_bf(A, K, row0, k0);
    v16bf b = load_B32x16_wt(W, K, k0, col0);
    acc = wmma_bf16(a, b, acc);
  }
  int lane = threadIdx.x & 31;
  int n  = col0 + (lane & 15);
  int mb = row0 + ((lane >> 4) << 3);
  float bv = bias ? bias[n] : 0.f;
#pragma unroll
  for (int r = 0; r < 8; ++r){
    float v = acc[r] + bv;
    if (relu) v = fmaxf(v, 0.f);
    C[(size_t)(mb + r) * N + n] = v;
    if (Cbf)  Cbf [(size_t)(mb + r) * N + n] = (bf16_t)v;
    if (CbfT) CbfT[(size_t)n * M + (mb + r)] = (bf16_t)v;
  }
}

// ---------------- one GRU timestep for both GRUs (h @ W_hh^T + gating), ping-pong H ----
__global__ void k_gru_step(const bf16_t* __restrict__ Hbf_in, const float* __restrict__ Hf_in,
                           bf16_t* Hbf_out, float* Hf_out,
                           const bf16_t* __restrict__ Whh0, const bf16_t* __restrict__ Whh1,
                           const float* __restrict__ bhh0, const float* __restrict__ bhh1,
                           const float* __restrict__ Gi0, const float* __restrict__ Gi1,
                           float* O0, float* O1, int t){
  int wave = (blockIdx.x * blockDim.x + threadIdx.x) >> 5;   // 256 waves
  int g = wave >> 7;
  int rem = wave & 127;
  int btile = rem >> 4, dtile = rem & 15;
  const bf16_t* Whh = g ? Whh1 : Whh0;
  const float*  bhh = g ? bhh1 : bhh0;
  const float*  Gi  = g ? Gi1  : Gi0;
  float* O = g ? O1 : O0;
  const bf16_t* Hb = Hbf_in + (size_t)g * BATCH * DIM;
  const float*  Hf = Hf_in  + (size_t)g * BATCH * DIM;
  bf16_t* Hbo = Hbf_out + (size_t)g * BATCH * DIM;
  float*  Hfo = Hf_out  + (size_t)g * BATCH * DIM;
  int row0 = btile << 4, col0 = dtile << 4;
  v8f ar = {}, az = {}, an = {};
  for (int k0 = 0; k0 < DIM; k0 += 32){
    v16bf a = load_A16x32_bf(Hb, DIM, row0, k0);
    ar = wmma_bf16(a, load_B32x16_wt(Whh, DIM, k0,           col0), ar);
    az = wmma_bf16(a, load_B32x16_wt(Whh, DIM, k0,   DIM  + col0), az);
    an = wmma_bf16(a, load_B32x16_wt(Whh, DIM, k0, 2*DIM  + col0), an);
  }
  int lane = threadIdx.x & 31;
  int dd = col0 + (lane & 15);
  int mb = row0 + ((lane >> 4) << 3);
#pragma unroll
  for (int r = 0; r < 8; ++r){
    int m = mb + r;
    size_t gi = ((size_t)m * TLEN + t) * G3D;
    float ghr = ar[r] + bhh[dd];
    float ghz = az[r] + bhh[DIM  + dd];
    float ghn = an[r] + bhh[2*DIM + dd];
    float rg = sigmoidf(Gi[gi + dd]          + ghr);
    float zg = sigmoidf(Gi[gi + DIM + dd]    + ghz);
    float ng = tanhf  (Gi[gi + 2*DIM + dd]   + rg * ghn);
    float ho = Hf[(size_t)m * DIM + dd];
    float hn = (1.f - zg) * ng + zg * ho;
    Hfo[(size_t)m * DIM + dd] = hn;
    Hbo[(size_t)m * DIM + dd] = (bf16_t)hn;
    O [((size_t)m * TLEN + t) * DIM + dd] = hn;
  }
}

// ---------------- query head: relu(concat(o1_last,o2_last)) @ query_w^T + query_b ------
__global__ void k_query(const float* __restrict__ O0, const float* __restrict__ O1,
                        const int* __restrict__ vlen, const float* __restrict__ qw,
                        const float* __restrict__ qb, float* query){
  int idx = blockIdx.x * blockDim.x + threadIdx.x;   // BATCH*DIM
  int b = idx >> 8, j = idx & 255;
  int t = vlen[b] - 1;
  const float* r1 = O0 + ((size_t)b * TLEN + t) * DIM;
  const float* r2 = O1 + ((size_t)b * TLEN + t) * DIM;
  const float* w  = qw + (size_t)j * (2*DIM);
  float acc = qb[j];
  for (int k = 0; k < DIM; ++k) acc += fmaxf(r1[k],0.f) * w[k];
  for (int k = 0; k < DIM; ++k) acc += fmaxf(r2[k],0.f) * w[DIM+k];
  query[idx] = acc;
}

// ---------------- GNN ----------------
__global__ void k_fp_gather(const int* __restrict__ fps, const float* __restrict__ fp_emb,
                            bf16_t* fvbf){
  int idx = blockIdx.x * blockDim.x + threadIdx.x;   // NA*DIM
  int a = idx >> 8, d = idx & 255;
  fvbf[idx] = (bf16_t)fp_emb[(size_t)fps[a]*DIM + d];
}

// fv_new = hs + adjacency @ hs ; each wave: one 16-row slab x 8 N-tiles (128 cols).
// B operand comes from the TRANSPOSED bf16 copy hsbfT [DIM x NA] so each lane's
// 16 K-values are contiguous (2x b128 loads per tile instead of 16 scalar loads).
__global__ void k_adj_gemm(const float* __restrict__ adj, const float* __restrict__ hs,
                           const bf16_t* __restrict__ hsbfT, float* fv, bf16_t* fvbf){
  int wave = (blockIdx.x * blockDim.x + threadIdx.x) >> 5;   // 1000 waves
  int mtile = wave >> 1;
  int ng    = wave & 1;
  int row0 = mtile << 4;
  int lane = threadIdx.x & 31;
  v8f zero = {};
  v8f acc[8];
#pragma unroll
  for (int j = 0; j < 8; ++j) acc[j] = zero;
  int mpref = row0 + (lane & 15);
  for (int k0 = 0; k0 < NAg; k0 += 32){
    if (k0 + 256 < NAg)
      __builtin_prefetch(adj + (size_t)mpref * NAg + k0 + 256, 0, 1);
    v16bf a = load_A16x32_f32(adj, NAg, row0, k0);
#pragma unroll
    for (int j = 0; j < 8; ++j){
      v16bf b = load_B32x16_wt(hsbfT, NAg, k0, (((ng<<3)+j) << 4));
      acc[j] = wmma_bf16(a, b, acc[j]);
    }
  }
  int mb = row0 + ((lane >> 4) << 3);
#pragma unroll
  for (int j = 0; j < 8; ++j){
    int n = (((ng<<3)+j) << 4) + (lane & 15);
#pragma unroll
    for (int r = 0; r < 8; ++r){
      float v = hs[(size_t)(mb + r) * DIM + n] + acc[j][r];
      fv  [(size_t)(mb + r) * DIM + n] = v;
      fvbf[(size_t)(mb + r) * DIM + n] = (bf16_t)v;
    }
  }
}

__global__ void k_seg_sum(const float* __restrict__ fv, const int* __restrict__ seg,
                          float* sums, float* cnts){
  int idx = blockIdx.x * blockDim.x + threadIdx.x;   // NA*DIM
  int a = idx >> 8, d = idx & 255;
  int m = seg[a];
  atomicAdd(&sums[(size_t)m*DIM + d], fv[idx]);
  if (d == 0) atomicAdd(&cnts[m], 1.f);
}
__global__ void k_seg_div(const float* __restrict__ sums, const float* __restrict__ cnts,
                          float* mol_vecs){
  int idx = blockIdx.x * blockDim.x + threadIdx.x;   // NMOL*DIM
  int m = idx >> 8;
  mol_vecs[idx] = sums[idx] / fmaxf(cnts[m], 1.f);
}
__global__ void k_mpnn_emb(const float* __restrict__ avg_proj, const float* __restrict__ mol_vecs,
                           float* mpnn_emb){
  int idx = blockIdx.x * blockDim.x + threadIdx.x;   // V2*DIM
  int i = idx >> 8, d = idx & 255;
  float acc = 0.f;
  for (int m = 0; m < NMOLg; ++m) acc += avg_proj[i*NMOLg + m] * mol_vecs[(size_t)m*DIM + d];
  mpnn_emb[idx] = acc;
}

// ---------------- epilogue ----------------
__global__ void k_match(const float* __restrict__ query, const float* __restrict__ mpnn_emb,
                        float* match){
  int idx = blockIdx.x * blockDim.x + threadIdx.x;   // BATCH*V2
  int b = idx / V2g, i = idx % V2g;
  float acc = 0.f;
  const float* q = query + (size_t)b*DIM;
  const float* e = mpnn_emb + (size_t)i*DIM;
  for (int k = 0; k < DIM; ++k) acc += q[k] * e[k];
  match[idx] = sigmoidf(acc);
}
__global__ void k_att_pre(const float* __restrict__ match, const float* __restrict__ out_w,
                          const float* __restrict__ out_b, float* pre){
  int idx = blockIdx.x * blockDim.x + threadIdx.x;   // BATCH*V2
  int b = idx / V2g, i = idx % V2g;
  float acc = match[idx] + out_b[i];
  const float* mr = match + (size_t)b*V2g;
  const float* wr = out_w + (size_t)i*V2g;
  for (int j = 0; j < V2g; ++j) acc += mr[j] * wr[j];
  pre[idx] = acc;
}
__global__ void k_lnorm(const float* __restrict__ pre, const float* __restrict__ g,
                        const float* __restrict__ bb, float* att){
  __shared__ float sdata[256];
  int b = blockIdx.x, tid = threadIdx.x;
  float x = (tid < V2g) ? pre[(size_t)b*V2g + tid] : 0.f;
  sdata[tid] = x;
  __syncthreads();
  for (int s = 128; s > 0; s >>= 1){ if (tid < s) sdata[tid] += sdata[tid+s]; __syncthreads(); }
  float mu = sdata[0] / (float)V2g;
  __syncthreads();
  float dv = (tid < V2g) ? (x - mu) : 0.f;
  sdata[tid] = dv*dv;
  __syncthreads();
  for (int s = 128; s > 0; s >>= 1){ if (tid < s) sdata[tid] += sdata[tid+s]; __syncthreads(); }
  float var = sdata[0] / (float)V2g;
  if (tid < V2g) att[(size_t)b*V2g + tid] = (x - mu) * rsqrtf(var + 1e-5f) * g[tid] + bb[tid];
}
__global__ void k_bip_in(const float* __restrict__ query, const float* __restrict__ bt_w,
                         const float* __restrict__ bt_b, float* bip_in){
  int idx = blockIdx.x * blockDim.x + threadIdx.x;   // BATCH*H1
  int b = idx / H1g, h = idx % H1g;
  float acc = bt_b[h];
  const float* q = query + (size_t)b*DIM;
  const float* w = bt_w + (size_t)h*DIM;
  for (int k = 0; k < DIM; ++k) acc += q[k] * w[k];
  bip_in[idx] = sigmoidf(acc);
}
__global__ void k_bip(const float* __restrict__ bip_in, const float* __restrict__ mask_w,
                      const float* __restrict__ ddi_mask_H, float* bip){
  int idx = blockIdx.x * blockDim.x + threadIdx.x;   // BATCH*V2
  int b = idx / V2g, i = idx % V2g;
  float acc = 0.f;
  const float* xr = bip_in + (size_t)b*H1g;
  for (int h = 0; h < H1g; ++h)
    acc += xr[h] * mask_w[(size_t)h*V2g + i] * ddi_mask_H[(size_t)i*H1g + h];
  bip[idx] = acc;
}
__global__ void k_result(const float* __restrict__ bip, const float* __restrict__ att, float* out){
  int idx = blockIdx.x * blockDim.x + threadIdx.x;   // BATCH*V2
  out[idx] = bip[idx] * att[idx];
}
__global__ void k_ddi(const float* __restrict__ res, const float* __restrict__ ddi,
                      float* out_scalar){
  __shared__ float negs[V2g];
  __shared__ float sdata[256];
  int b = blockIdx.x, tid = threadIdx.x;
  if (tid < V2g) negs[tid] = sigmoidf(res[(size_t)b*V2g + tid]);
  __syncthreads();
  float p = 0.f;
  if (tid < V2g){
    float s = 0.f;
    const float* dr = ddi + (size_t)tid * V2g;
    for (int j = 0; j < V2g; ++j) s += dr[j] * negs[j];
    p = negs[tid] * s;
  }
  sdata[tid] = p;
  __syncthreads();
  for (int s = 128; s > 0; s >>= 1){ if (tid < s) sdata[tid] += sdata[tid+s]; __syncthreads(); }
  if (tid == 0) atomicAdd(out_scalar, 0.0005f * sdata[0]);
}

// ---------------- host launcher ----------------
extern "C" void kernel_launch(void* const* d_in, const int* in_sizes, int n_in,
                              void* d_out, int out_size, void* d_ws, size_t ws_size,
                              hipStream_t stream) {
  const int*   diag   = (const int*)  d_in[0];
  const int*   pro    = (const int*)  d_in[1];
  const int*   vlen   = (const int*)  d_in[2];
  const int*   fps    = (const int*)  d_in[3];
  const int*   segid  = (const int*)  d_in[4];
  const float* adj    = (const float*)d_in[5];
  const float* emb0   = (const float*)d_in[6];
  const float* emb1   = (const float*)d_in[7];
  const float* g1_wih = (const float*)d_in[8];
  const float* g1_whh = (const float*)d_in[9];
  const float* g1_bih = (const float*)d_in[10];
  const float* g1_bhh = (const float*)d_in[11];
  const float* g2_wih = (const float*)d_in[12];
  const float* g2_whh = (const float*)d_in[13];
  const float* g2_bih = (const float*)d_in[14];
  const float* g2_bhh = (const float*)d_in[15];
  const float* qw     = (const float*)d_in[16];
  const float* qb     = (const float*)d_in[17];
  const float* bt_w   = (const float*)d_in[18];
  const float* bt_b   = (const float*)d_in[19];
  const float* mask_w = (const float*)d_in[20];
  const float* fp_emb = (const float*)d_in[21];
  const float* fpw0   = (const float*)d_in[22];
  const float* fpb0   = (const float*)d_in[23];
  const float* fpw1   = (const float*)d_in[24];
  const float* fpb1   = (const float*)d_in[25];
  const float* avg_pj = (const float*)d_in[26];
  const float* out_w  = (const float*)d_in[27];
  const float* out_b  = (const float*)d_in[28];
  const float* ln_g   = (const float*)d_in[29];
  const float* ln_b   = (const float*)d_in[30];
  const float* ddimH  = (const float*)d_in[31];
  const float* ddiadj = (const float*)d_in[32];
  float* out = (float*)d_out;

  char* base = (char*)d_ws;
  size_t off = 0;
  auto alloc = [&](size_t bytes)->void*{ off = (off + 255) & ~(size_t)255; void* p = base + off; off += bytes; return p; };

  const size_t BT = (size_t)BATCH * TLEN;            // 4096
  bf16_t* X0bf   = (bf16_t*)alloc(BT*DIM*2);
  bf16_t* X1bf   = (bf16_t*)alloc(BT*DIM*2);
  bf16_t* wih1bf = (bf16_t*)alloc((size_t)G3D*DIM*2);
  bf16_t* whh1bf = (bf16_t*)alloc((size_t)G3D*DIM*2);
  bf16_t* wih2bf = (bf16_t*)alloc((size_t)G3D*DIM*2);
  bf16_t* whh2bf = (bf16_t*)alloc((size_t)G3D*DIM*2);
  bf16_t* fpw0bf = (bf16_t*)alloc((size_t)DIM*DIM*2);
  bf16_t* fpw1bf = (bf16_t*)alloc((size_t)DIM*DIM*2);
  float*  Gi0    = (float*) alloc(BT*G3D*4);
  float*  Gi1    = (float*) alloc(BT*G3D*4);
  float*  Hf[2]; bf16_t* Hb[2];
  Hf[0] = (float*) alloc((size_t)2*BATCH*DIM*4);
  Hf[1] = (float*) alloc((size_t)2*BATCH*DIM*4);
  Hb[0] = (bf16_t*)alloc((size_t)2*BATCH*DIM*2);
  Hb[1] = (bf16_t*)alloc((size_t)2*BATCH*DIM*2);
  float*  O0     = (float*) alloc(BT*DIM*4);
  float*  O1     = (float*) alloc(BT*DIM*4);
  float*  query  = (float*) alloc((size_t)BATCH*DIM*4);
  bf16_t* fvbf   = (bf16_t*)alloc((size_t)NAg*DIM*2);
  float*  hs     = (float*) alloc((size_t)NAg*DIM*4);
  bf16_t* hsbfT  = (bf16_t*)alloc((size_t)NAg*DIM*2);   // transposed [DIM x NA]
  float*  fv     = (float*) alloc((size_t)NAg*DIM*4);
  float*  sums   = (float*) alloc((size_t)NMOLg*DIM*4);
  float*  cnts   = (float*) alloc((size_t)NMOLg*4);
  float*  molv   = (float*) alloc((size_t)NMOLg*DIM*4);
  float*  memb   = (float*) alloc((size_t)V2g*DIM*4);
  float*  match  = (float*) alloc((size_t)BATCH*V2g*4);
  float*  pre    = (float*) alloc((size_t)BATCH*V2g*4);
  float*  att    = (float*) alloc((size_t)BATCH*V2g*4);
  float*  bipin  = (float*) alloc((size_t)BATCH*H1g*4);
  float*  bip    = (float*) alloc((size_t)BATCH*V2g*4);
  (void)ws_size; (void)n_in; (void)in_sizes; (void)out_size;

  // 1) pooled embeddings -> bf16 GRU inputs
  k_embed<<<dim3(4096,2), 256, 0, stream>>>(diag, pro, emb0, emb1, X0bf, X1bf);

  // 2) weight conversions to bf16
  k_f32_to_bf16<<<768, 256, 0, stream>>>(g1_wih, wih1bf, G3D*DIM);
  k_f32_to_bf16<<<768, 256, 0, stream>>>(g1_whh, whh1bf, G3D*DIM);
  k_f32_to_bf16<<<768, 256, 0, stream>>>(g2_wih, wih2bf, G3D*DIM);
  k_f32_to_bf16<<<768, 256, 0, stream>>>(g2_whh, whh2bf, G3D*DIM);
  k_f32_to_bf16<<<256, 256, 0, stream>>>(fpw0, fpw0bf, DIM*DIM);
  k_f32_to_bf16<<<256, 256, 0, stream>>>(fpw1, fpw1bf, DIM*DIM);

  // 3) zero initial hidden state (both f32 + bf16 copies; bf16 zero == bits zero)
  k_zero_f32<<<256, 256, 0, stream>>>(Hf[0], 2*BATCH*DIM);
  k_zero_f32<<<128, 256, 0, stream>>>((float*)Hb[0], (2*BATCH*DIM)/2);

  // 4) Gi = X @ W_ih^T + b_ih for both GRUs (WMMA): M=4096,N=768,K=256 -> 12288 waves
  k_gemm_wt<<<1536, 256, 0, stream>>>(X0bf, wih1bf, g1_bih, Gi0,
                                      (bf16_t*)nullptr, (bf16_t*)nullptr, BT, G3D, DIM, 0);
  k_gemm_wt<<<1536, 256, 0, stream>>>(X1bf, wih2bf, g2_bih, Gi1,
                                      (bf16_t*)nullptr, (bf16_t*)nullptr, BT, G3D, DIM, 0);

  // 5) 32 recurrent steps, ping-pong H (256 waves/step)
  for (int t = 0; t < TLEN; ++t){
    int src = t & 1, dst = src ^ 1;
    k_gru_step<<<32, 256, 0, stream>>>(Hb[src], Hf[src], Hb[dst], Hf[dst],
                                       whh1bf, whh2bf, g1_bhh, g2_bhh,
                                       Gi0, Gi1, O0, O1, t);
  }

  // 6) query head
  k_query<<<128, 256, 0, stream>>>(O0, O1, vlen, qw, qb, query);

  // 7) GNN: gather fingerprints, 2 layers of (linear+relu, adjacency aggregate)
  k_fp_gather<<<8000, 256, 0, stream>>>(fps, fp_emb, fvbf);
  k_gemm_wt<<<1000, 256, 0, stream>>>(fvbf, fpw0bf, fpb0, hs,
                                      (bf16_t*)nullptr, hsbfT, NAg, DIM, DIM, 1);
  k_adj_gemm<<<125, 256, 0, stream>>>(adj, hs, hsbfT, fv, fvbf);
  k_gemm_wt<<<1000, 256, 0, stream>>>(fvbf, fpw1bf, fpb1, hs,
                                      (bf16_t*)nullptr, hsbfT, NAg, DIM, DIM, 1);
  k_adj_gemm<<<125, 256, 0, stream>>>(adj, hs, hsbfT, fv, fvbf);

  // 8) segment mean -> mol vecs -> mpnn embedding
  k_zero_f32<<<256, 256, 0, stream>>>(sums, NMOLg*DIM);
  k_zero_f32<<<1, 256, 0, stream>>>(cnts, NMOLg);
  k_seg_sum<<<8000, 256, 0, stream>>>(fv, segid, sums, cnts);
  k_seg_div<<<250, 256, 0, stream>>>(sums, cnts, molv);
  k_mpnn_emb<<<200, 256, 0, stream>>>(avg_pj, molv, memb);

  // 9) epilogue
  k_match<<<100, 256, 0, stream>>>(query, memb, match);
  k_att_pre<<<100, 256, 0, stream>>>(match, out_w, out_b, pre);
  k_lnorm<<<128, 256, 0, stream>>>(pre, ln_g, ln_b, att);
  k_bip_in<<<250, 256, 0, stream>>>(query, bt_w, bt_b, bipin);
  k_bip<<<100, 256, 0, stream>>>(bipin, mask_w, ddimH, bip);
  k_result<<<100, 256, 0, stream>>>(bip, att, out);

  // 10) batch_neg scalar (zero then accumulate)
  k_zero_f32<<<1, 256, 0, stream>>>(out + (size_t)BATCH*V2g, 1);
  k_ddi<<<128, 256, 0, stream>>>(out, ddiadj, out + (size_t)BATCH*V2g);
}
